// StarTransformerNER_75960791597417
// MI455X (gfx1250) — matrix operand order repaired and checked by reading, not compiled, not run on hardware
//
#include <hip/hip_runtime.h>
#include <hip/hip_bf16.h>

typedef __attribute__((ext_vector_type(16))) _Float16 v16h;
typedef __attribute__((ext_vector_type(8)))  float    v8f;

#define DEVINL __device__ __forceinline__

namespace {

constexpr int B_   = 16, L_ = 512, H_ = 768, NL_ = 12, DH_ = 64, AH_ = 768;
constexpr int TOK  = B_ * L_;        // 8192 tokens per pass
constexpr int ROWS2 = 2 * TOK;       // 16384 (fwd + bwd stacked)
constexpr int KT_H = H_ / 32;        // 24 K-tiles over hidden dim
constexpr int NT_AH = AH_ / 16;      // 48 N-tiles over all-head dim
constexpr int MT2  = ROWS2 / 16;     // 1024 M-tiles (both passes)
constexpr int PBH  = 2 * B_ * NL_;   // 384 (pass, batch, head) combos
constexpr int PAD  = 800;            // 769 padded up to 32/16 multiple
constexpr int KT_P = PAD / 32;       // 25
constexpr int NT_P = PAD / 16;       // 50
constexpr int FR   = 512;            // halves per fragment (32 lanes x 16)

// ---- WMMA fragment index maps (CDNA5 ISA 7.12.2, f16, wave32) ----
// A 16x32: lane l -> m = l&15, hi = l>>4; element e -> K
DEVINL int aK(int e, int hi) {
  int p = e >> 1, lo = e & 1;
  return (p < 4 ? 2 * p : 16 + 2 * (p - 4)) + 8 * hi + lo;
}
// B 32x16: lane l -> n = l&15, hi = l>>4; element e -> K
DEVINL int bK(int e, int hi) {
  int p = e >> 1, lo = e & 1;
  return (p < 4 ? 2 * p : 16 + 2 * (p - 4)) + 8 * hi + lo;
}

DEVINL v8f wmma(v16h a, v16h b, v8f c) {
  return __builtin_amdgcn_wmma_f32_16x16x32_f16(false, a, false, b, (short)0, c,
                                                false, false);
}
DEVINL v16h ldfrag(const _Float16* p) { return *(const v16h*)p; }
DEVINL void stfrag(_Float16* p, v16h v) { *(v16h*)p = v; }

// ---- CDNA5 async global->LDS copy (ASYNCcnt-tracked, §10 VGLOBAL async) ----
DEVINL void async_b128(unsigned lds_off, const void* gaddr) {
  asm volatile("global_load_async_to_lds_b128 %0, %1, off"
               :: "v"(lds_off), "v"(gaddr) : "memory");
}
DEVINL void wait_async_le8() { asm volatile("s_wait_asynccnt 0x8" ::: "memory"); }
DEVINL void wait_async_0()   { asm volatile("s_wait_asynccnt 0x0" ::: "memory"); }

// ============================================================
// 1) Embedding gather + cycle-shift -> X fragments (A-layout, f16)
//    rows 0..8191 = fwd pass, rows 8192..16383 = bwd pass
// ============================================================
__global__ void k_pack_x(const int* __restrict__ ids,
                         const float* __restrict__ emb,
                         _Float16* __restrict__ xfrag) {
  int tid = blockIdx.x * blockDim.x + threadIdx.x;
  if (tid >= MT2 * KT_H * 32) return;
  int lane = tid & 31; int t = tid >> 5;
  int tk = t % KT_H, tm = t / KT_H;
  int hi = lane >> 4;
  int row = 16 * tm + (lane & 15);
  int pass = row >> 13; int tt = row & (TOK - 1);
  int b = tt >> 9, i = tt & (L_ - 1);
  // fwd: pos0 <- last token, pos1.. <- token0 ; bwd: roll left by 1
  int src = (pass == 0) ? (i == 0 ? L_ - 1 : 0) : ((i + 1) & (L_ - 1));
  const float* e = emb + (size_t)ids[b * L_ + src] * H_;
  v16h v;
#pragma unroll
  for (int q = 0; q < 16; ++q) v[q] = (_Float16)e[32 * tk + aK(q, hi)];
  stfrag(xfrag + ((size_t)(tm * KT_H + tk)) * FR + lane * 16, v);
}

// ============================================================
// 2) Weight pack: W[768x768] row-major -> B fragments
// ============================================================
__global__ void k_pack_w(const float* __restrict__ W,
                         _Float16* __restrict__ wfrag) {
  int tid = blockIdx.x * blockDim.x + threadIdx.x;
  if (tid >= KT_H * NT_AH * 32) return;
  int lane = tid & 31; int t = tid >> 5;
  int tn = t % NT_AH, tk = t / NT_AH;
  int hi = lane >> 4;
  int n = 16 * tn + (lane & 15);
  v16h v;
#pragma unroll
  for (int q = 0; q < 16; ++q)
    v[q] = (_Float16)W[(size_t)(32 * tk + bK(q, hi)) * AH_ + n];
  stfrag(wfrag + ((size_t)(tk * NT_AH + tn)) * FR + lane * 16, v);
}

// ============================================================
// 3) QKV GEMM: [16384 x 768] @ [768 x 768] + bias, one wave = 32x64 tile
// ============================================================
__global__ void __launch_bounds__(32)
k_qkv(const _Float16* __restrict__ xfrag, const _Float16* __restrict__ wq,
      const _Float16* __restrict__ wk, const _Float16* __restrict__ wv,
      const float* __restrict__ bq, const float* __restrict__ bk,
      const float* __restrict__ bv, float* __restrict__ qkv) {
  int id = blockIdx.x, lane = threadIdx.x;
  int ng = id % 12; int t = id / 12;
  int mat = t % 3, tm2 = t / 3;  // tm2 < 512, covers m-tiles 2*tm2, 2*tm2+1
  const _Float16* wf = (mat == 0) ? wq : (mat == 1) ? wk : wv;
  const float* bias = (mat == 0) ? bq : (mat == 1) ? bk : bv;
  v8f acc[2][4] = {};
  for (int tk = 0; tk < KT_H; ++tk) {
    const _Float16* ap0 =
        xfrag + ((size_t)((2 * tm2 + 0) * KT_H + tk)) * FR + lane * 16;
    const _Float16* ap1 =
        xfrag + ((size_t)((2 * tm2 + 1) * KT_H + tk)) * FR + lane * 16;
    __builtin_prefetch(ap0 + FR, 0, 1);
    v16h a0 = ldfrag(ap0), a1 = ldfrag(ap1);
#pragma unroll
    for (int jj = 0; jj < 4; ++jj) {
      v16h bfr = ldfrag(wf + ((size_t)(tk * NT_AH + ng * 4 + jj)) * FR + lane * 16);
      acc[0][jj] = wmma(a0, bfr, acc[0][jj]);
      acc[1][jj] = wmma(a1, bfr, acc[1][jj]);
    }
  }
  int hi = lane >> 4, nl = lane & 15;
  size_t base = (size_t)mat * ROWS2 * AH_;
#pragma unroll
  for (int as = 0; as < 2; ++as)
#pragma unroll
    for (int jj = 0; jj < 4; ++jj) {
      int n = (ng * 4 + jj) * 16 + nl;
      float bb = bias[n];
#pragma unroll
      for (int r = 0; r < 8; ++r) {
        int row = 16 * (2 * tm2 + as) + r + 8 * hi;
        qkv[base + (size_t)row * AH_ + n] = acc[as][jj][r] + bb;
      }
    }
}

// ---- interleaved RoPE (matches reference: repeat(cos,2), pair mixing) ----
DEVINL float rope_val(const float* __restrict__ head, int d, int pos) {
  float x0 = head[d], xn = head[d ^ 1];
  int pr = d >> 1;
  float ang = (float)pos * __expf(-(float)pr * 0.28782313662425575f); // ln(1e4)*2/64
  float c = __cosf(ang), s = __sinf(ang);
  return (d & 1) ? (x0 * c + xn * s) : (x0 * c - xn * s);
}

// ============================================================
// 4a) Q -> A fragments (per pass,b,h; M=512, K=64), RoPE fused
// ============================================================
__global__ void k_pack_q(const float* __restrict__ qkv,
                         _Float16* __restrict__ qfrag) {
  int tid = blockIdx.x * blockDim.x + threadIdx.x;
  if (tid >= PBH * 32 * 2 * 32) return;
  int lane = tid & 31; int t = tid >> 5;
  int tk = t & 1; t >>= 1;
  int tm = t & 31; t >>= 5;
  int pbh = t;
  int h = pbh % 12; int bp = pbh / 12; int b = bp & 15; int pass = bp >> 4;
  int hi = lane >> 4, m = lane & 15;
  int x = 16 * tm + m;
  const float* head = qkv + (size_t)(pass * TOK + b * L_ + x) * AH_ + h * DH_;
  v16h v;
#pragma unroll
  for (int q = 0; q < 16; ++q)
    v[q] = (_Float16)rope_val(head, 32 * tk + aK(q, hi), x);
  stfrag(qfrag + ((size_t)pbh * 64 + tm * 2 + tk) * FR + lane * 16, v);
}

// ============================================================
// 4b) K^T -> B fragments (scores GEMM: Kdim=DH=64, N=512), RoPE fused
// ============================================================
__global__ void k_pack_k(const float* __restrict__ qkv,
                         _Float16* __restrict__ kfrag) {
  int tid = blockIdx.x * blockDim.x + threadIdx.x;
  if (tid >= PBH * 32 * 2 * 32) return;
  int lane = tid & 31; int t = tid >> 5;
  int tk = t & 1; t >>= 1;
  int tn = t & 31; t >>= 5;
  int pbh = t;
  int h = pbh % 12; int bp = pbh / 12; int b = bp & 15; int pass = bp >> 4;
  int hi = lane >> 4;
  int y = 16 * tn + (lane & 15);
  const float* head = qkv + ((size_t)ROWS2 * AH_) +
                      (size_t)(pass * TOK + b * L_ + y) * AH_ + h * DH_;
  v16h v;
#pragma unroll
  for (int q = 0; q < 16; ++q)
    v[q] = (_Float16)rope_val(head, 32 * tk + bK(q, hi), y);
  stfrag(kfrag + ((size_t)pbh * 64 + tn * 2 + tk) * FR + lane * 16, v);
}

// ============================================================
// 4c) V -> B fragments (ctx GEMM: Kdim=512, N=64)
// ============================================================
__global__ void k_pack_v(const float* __restrict__ qkv,
                         _Float16* __restrict__ vfrag) {
  int tid = blockIdx.x * blockDim.x + threadIdx.x;
  if (tid >= PBH * 16 * 4 * 32) return;
  int lane = tid & 31; int t = tid >> 5;
  int tn = t & 3; t >>= 2;
  int tk = t & 15; t >>= 4;
  int pbh = t;
  int h = pbh % 12; int bp = pbh / 12; int b = bp & 15; int pass = bp >> 4;
  int hi = lane >> 4;
  int dc = 16 * tn + (lane & 15);
  const float* vsrc = qkv + 2 * (size_t)ROWS2 * AH_;
  v16h v;
#pragma unroll
  for (int q = 0; q < 16; ++q) {
    int y = 32 * tk + bK(q, hi);
    v[q] = (_Float16)vsrc[(size_t)(pass * TOK + b * L_ + y) * AH_ + h * DH_ + dc];
  }
  stfrag(vfrag + ((size_t)pbh * 64 + tk * 4 + tn) * FR + lane * 16, v);
}

// ============================================================
// 5) Attention: one wave = (pass,b,h, 16-row q block)
//    scores (WMMA) -> LDS -> softmax -> ctx (WMMA)
// ============================================================
__global__ void __launch_bounds__(32)
k_attn(const _Float16* __restrict__ qfrag, const _Float16* __restrict__ kfrag,
       const _Float16* __restrict__ vfrag, float* __restrict__ ctx) {
  __shared__ float S[16][520];      // padded row stride vs bank conflicts
  __shared__ float red[2][16][2];
  __shared__ float invs[16];
  int lane = threadIdx.x;
  int qt = blockIdx.x & 31, pbh = blockIdx.x >> 5;
  int h = pbh % 12; int bp = pbh / 12; int b = bp & 15; int pass = bp >> 4;
  int hi = lane >> 4, nl = lane & 15;
  const float scale = 0.03608439182435161f;  // 1/sqrt(768) per reference

  v16h qa0 = ldfrag(qfrag + ((size_t)pbh * 64 + qt * 2 + 0) * FR + lane * 16);
  v16h qa1 = ldfrag(qfrag + ((size_t)pbh * 64 + qt * 2 + 1) * FR + lane * 16);

  for (int tn = 0; tn < 32; ++tn) {
    v8f acc = {};
    acc = wmma(qa0, ldfrag(kfrag + ((size_t)pbh * 64 + tn * 2 + 0) * FR + lane * 16), acc);
    acc = wmma(qa1, ldfrag(kfrag + ((size_t)pbh * 64 + tn * 2 + 1) * FR + lane * 16), acc);
    int n = 16 * tn + nl;
#pragma unroll
    for (int r = 0; r < 8; ++r) S[r + 8 * hi][n] = acc[r] * scale;
  }
  __syncthreads();

  // softmax: lane (nl,hi) handles row nl, column segment hi*256..+255
  float mx = -1e30f;
  for (int c = 0; c < 256; ++c) mx = fmaxf(mx, S[nl][hi * 256 + c]);
  red[0][nl][hi] = mx;
  __syncthreads();
  float m = fmaxf(red[0][nl][0], red[0][nl][1]);
  float sum = 0.f;
  for (int c = 0; c < 256; ++c) {
    float e = __expf(S[nl][hi * 256 + c] - m);
    S[nl][hi * 256 + c] = e;
    sum += e;
  }
  red[1][nl][hi] = sum;
  __syncthreads();
  if (hi == 0) invs[nl] = 1.0f / (red[1][nl][0] + red[1][nl][1]);
  __syncthreads();

  // ctx = P[16x512] @ V[512x64]; P fragments built from LDS with 1/sum folded in
  v8f c[4] = {};
  float is = invs[nl];
  for (int tk = 0; tk < 16; ++tk) {
    v16h pa;
#pragma unroll
    for (int q = 0; q < 16; ++q)
      pa[q] = (_Float16)(S[nl][32 * tk + aK(q, hi)] * is);
#pragma unroll
    for (int tn = 0; tn < 4; ++tn)
      c[tn] = wmma(pa, ldfrag(vfrag + ((size_t)pbh * 64 + tk * 4 + tn) * FR + lane * 16), c[tn]);
  }
  int rowbase = pass * TOK + b * L_ + 16 * qt;
#pragma unroll
  for (int tn = 0; tn < 4; ++tn)
#pragma unroll
    for (int r = 0; r < 8; ++r)
      ctx[(size_t)(rowbase + r + 8 * hi) * AH_ + h * DH_ + 16 * tn + nl] = c[tn][r];
}

// ============================================================
// 6) LayerNorm + append-ones + zero-pad to 800, f16 rows
// ============================================================
__global__ void k_ln(const float* __restrict__ ctx, const float* __restrict__ lnw,
                     const float* __restrict__ lnb, _Float16* __restrict__ xf,
                     _Float16* __restrict__ yb) {
  int row = blockIdx.x, tid = threadIdx.x;
  const float* src = ctx + (size_t)row * AH_;
  __shared__ float s1[256], s2[256];
  float a = 0.f, sq = 0.f;
  for (int c = tid; c < AH_; c += 256) { float v = src[c]; a += v; sq += v * v; }
  s1[tid] = a; s2[tid] = sq;
  __syncthreads();
  for (int off = 128; off; off >>= 1) {
    if (tid < off) { s1[tid] += s1[tid + off]; s2[tid] += s2[tid + off]; }
    __syncthreads();
  }
  float mu = s1[0] * (1.0f / AH_);
  float var = s2[0] * (1.0f / AH_) - mu * mu;
  float rstd = rsqrtf(var + 1e-12f);
  int pass = row >> 13, token = row & (TOK - 1);
  _Float16* dst = (pass ? yb : xf) + (size_t)token * PAD;
  for (int c = tid; c < PAD; c += 256) {
    float v = (c < AH_) ? (lnw[c] * (src[c] - mu) * rstd + lnb[c])
                        : (c == AH_ ? 1.0f : 0.0f);
    dst[c] = (_Float16)v;
  }
}

// ============================================================
// 7) xf rows -> A fragments ([8192 x 800])
// ============================================================
__global__ void k_pack_xf(const _Float16* __restrict__ src,
                          _Float16* __restrict__ frag) {
  int tid = blockIdx.x * blockDim.x + threadIdx.x;
  if (tid >= (TOK / 16) * KT_P * 32) return;
  int lane = tid & 31; int t = tid >> 5;
  int tk = t % KT_P, tm = t / KT_P;
  int hi = lane >> 4;
  const _Float16* s = src + (size_t)(16 * tm + (lane & 15)) * PAD;
  v16h v;
#pragma unroll
  for (int q = 0; q < 16; ++q) v[q] = s[32 * tk + aK(q, hi)];
  stfrag(frag + ((size_t)(tm * KT_P + tk)) * FR + lane * 16, v);
}

// ============================================================
// 8) U[769,12,769] -> per-o B fragments ([800 x 800] zero padded)
// ============================================================
__global__ void k_pack_u(const float* __restrict__ U,
                         _Float16* __restrict__ ufrag) {
  int tid = blockIdx.x * blockDim.x + threadIdx.x;
  if (tid >= NL_ * KT_P * NT_P * 32) return;
  int lane = tid & 31; int t = tid >> 5;
  int tn = t % NT_P; t /= NT_P;
  int tk = t % KT_P; int o = t / KT_P;
  int hi = lane >> 4;
  int j = 16 * tn + (lane & 15);
  v16h v;
#pragma unroll
  for (int q = 0; q < 16; ++q) {
    int i = 32 * tk + bK(q, hi);
    v[q] = (i < 769 && j < 769) ? (_Float16)U[((size_t)i * NL_ + o) * 769 + j]
                                : (_Float16)0.0f;
  }
  stfrag(ufrag + ((size_t)((o * KT_P + tk) * NT_P + tn)) * FR + lane * 16, v);
}

// ============================================================
// 9) Bilinear stage 1: T[b,x,o,j] = sum_i xf[b,x,i] U[i,o,j]
//    one wave = 32x32 output tile (2 A-frags x 2 B-frags)
// ============================================================
__global__ void __launch_bounds__(32)
k_bil1(const _Float16* __restrict__ xffrag, const _Float16* __restrict__ ufrag,
       _Float16* __restrict__ T) {
  int id = blockIdx.x, lane = threadIdx.x;
  int ng = id % 25; int t = id / 25;
  int o = t % 12, tm2 = t / 12;  // tm2 < 256
  v8f acc[2][2] = {};
  for (int tk = 0; tk < KT_P; ++tk) {
    const _Float16* ap0 =
        xffrag + ((size_t)((2 * tm2 + 0) * KT_P + tk)) * FR + lane * 16;
    const _Float16* ap1 =
        xffrag + ((size_t)((2 * tm2 + 1) * KT_P + tk)) * FR + lane * 16;
    __builtin_prefetch(ap0 + FR, 0, 1);
    v16h a0 = ldfrag(ap0), a1 = ldfrag(ap1);
#pragma unroll
    for (int jj = 0; jj < 2; ++jj) {
      v16h bfr = ldfrag(ufrag + ((size_t)((o * KT_P + tk) * NT_P + ng * 2 + jj)) * FR + lane * 16);
      acc[0][jj] = wmma(a0, bfr, acc[0][jj]);
      acc[1][jj] = wmma(a1, bfr, acc[1][jj]);
    }
  }
  int hi = lane >> 4, nl = lane & 15;
#pragma unroll
  for (int as = 0; as < 2; ++as)
#pragma unroll
    for (int jj = 0; jj < 2; ++jj) {
      int j = 16 * (ng * 2 + jj) + nl;
#pragma unroll
      for (int r = 0; r < 8; ++r) {
        int token = 16 * (2 * tm2 + as) + r + 8 * hi;
        T[((size_t)token * NL_ + o) * PAD + j] = (_Float16)acc[as][jj][r];
      }
    }
}

// ============================================================
// 10) T -> per-(b,o) A fragments (M=512, K=800)
// ============================================================
__global__ void k_pack_t(const _Float16* __restrict__ T,
                         _Float16* __restrict__ tfrag) {
  int tid = blockIdx.x * blockDim.x + threadIdx.x;
  if (tid >= B_ * NL_ * 32 * KT_P * 32) return;
  int lane = tid & 31; int t = tid >> 5;
  int tk = t % KT_P; t /= KT_P;
  int tm = t & 31; t >>= 5;
  int o = t % 12, b = t / 12;
  int hi = lane >> 4;
  int x = 16 * tm + (lane & 15);
  const _Float16* s = T + ((size_t)(b * L_ + x) * NL_ + o) * PAD;
  v16h v;
#pragma unroll
  for (int q = 0; q < 16; ++q) v[q] = s[32 * tk + aK(q, hi)];
  stfrag(tfrag + ((size_t)(((b * 12 + o) * 32 + tm) * KT_P + tk)) * FR + lane * 16, v);
}

// ============================================================
// 11) yb -> per-b B fragments of yb^T (Kdim=j=800, N=y=512)
// ============================================================
__global__ void k_pack_yb(const _Float16* __restrict__ ybuf,
                          _Float16* __restrict__ ybfrag) {
  int tid = blockIdx.x * blockDim.x + threadIdx.x;
  if (tid >= B_ * KT_P * 32 * 32) return;
  int lane = tid & 31; int t = tid >> 5;
  int tn = t & 31; t >>= 5;
  int tk = t % KT_P; int b = t / KT_P;
  int hi = lane >> 4;
  int y = 16 * tn + (lane & 15);
  const _Float16* s = ybuf + (size_t)(b * L_ + y) * PAD;
  v16h v;
#pragma unroll
  for (int q = 0; q < 16; ++q) v[q] = s[32 * tk + bK(q, hi)];
  stfrag(ybfrag + ((size_t)((b * KT_P + tk) * 32 + tn)) * FR + lane * 16, v);
}

// ============================================================
// 12) Bilinear stage 2: out[b,x,y,o] = sum_j T[b,x,o,j] yb[b,y,j]
//     4-wave workgroup = 64x64 block tile; the 4 shared yb B-fragments
//     are double-buffered in LDS via global_load_async_to_lds_b128.
// ============================================================
__global__ void __launch_bounds__(128)
k_bil2(const _Float16* __restrict__ tfrag, const _Float16* __restrict__ ybfrag,
       float* __restrict__ out) {
  __shared__ __align__(32) _Float16 Bs[2][4][FR];
  int lane = threadIdx.x & 31, wv = threadIdx.x >> 5;
  int id = blockIdx.x;
  int ng = id & 7; int t = id >> 3;
  int mg = t & 7; t >>= 3;
  int o = t % 12, b = t / 12;
  int tm = mg * 4 + wv;  // this wave's m-tile (16 rows)

  // wave 0 stages the 4 shared B fragments (4 KB) for one K-step
  auto stage = [&](int tk, int p) {
    const _Float16* g =
        ybfrag + ((size_t)((b * KT_P + tk) * 32 + ng * 4)) * FR + lane * 16;
#pragma unroll
    for (int f = 0; f < 4; ++f) {
      unsigned l = (unsigned)(uintptr_t)(&Bs[p][f][0]) + (unsigned)lane * 32u;
      const char* gp = (const char*)(g + (size_t)f * FR);
      async_b128(l, gp);
      async_b128(l + 16u, gp + 16);
    }
  };

  if (wv == 0) stage(0, 0);
  v8f acc[4] = {};
  for (int tk = 0; tk < KT_P; ++tk) {
    int p = tk & 1;
    if (wv == 0) {
      if (tk + 1 < KT_P) { stage(tk + 1, p ^ 1); wait_async_le8(); }
      else               { wait_async_0(); }
    }
    __syncthreads();  // Bs[p] ready for all waves
    const _Float16* ap =
        tfrag + ((size_t)(((b * 12 + o) * 32 + tm) * KT_P + tk)) * FR + lane * 16;
    __builtin_prefetch(ap + FR, 0, 1);
    v16h a = ldfrag(ap);
#pragma unroll
    for (int jj = 0; jj < 4; ++jj)
      acc[jj] = wmma(a, ldfrag(&Bs[p][jj][lane * 16]), acc[jj]);
    __syncthreads();  // Bs[p] fully consumed before it is refilled
  }
  int hi = lane >> 4, nl = lane & 15;
#pragma unroll
  for (int jj = 0; jj < 4; ++jj) {
    int y = 16 * (ng * 4 + jj) + nl;
#pragma unroll
    for (int r = 0; r < 8; ++r) {
      int x = 16 * tm + r + 8 * hi;
      out[((size_t)(b * L_ + x) * L_ + y) * NL_ + o] = acc[jj][r];
    }
  }
}

}  // namespace

extern "C" void kernel_launch(void* const* d_in, const int* in_sizes, int n_in,
                              void* d_out, int out_size, void* d_ws, size_t ws_size,
                              hipStream_t stream) {
  const int*   ids  = (const int*)d_in[0];
  const float* emb  = (const float*)d_in[1];
  const float* Wq   = (const float*)d_in[2];
  const float* bq   = (const float*)d_in[3];
  const float* Wk   = (const float*)d_in[4];
  const float* bk   = (const float*)d_in[5];
  const float* Wv   = (const float*)d_in[6];
  const float* bv   = (const float*)d_in[7];
  const float* lnw  = (const float*)d_in[8];
  const float* lnb  = (const float*)d_in[9];
  const float* U    = (const float*)d_in[10];
  float* out = (float*)d_out;
  char* ws = (char*)d_ws;

  size_t cur = 0;
  auto alloc = [&](size_t bytes) -> char* {
    char* p = ws + cur;
    cur += (bytes + 255) & ~(size_t)255;
    return p;
  };
  _Float16* xfrag  = (_Float16*)alloc((size_t)MT2 * KT_H * FR * 2);
  _Float16* wfq    = (_Float16*)alloc((size_t)KT_H * NT_AH * FR * 2);
  _Float16* wfk    = (_Float16*)alloc((size_t)KT_H * NT_AH * FR * 2);
  _Float16* wfv    = (_Float16*)alloc((size_t)KT_H * NT_AH * FR * 2);
  float*    qkv    = (float*)   alloc((size_t)3 * ROWS2 * AH_ * 4);
  _Float16* qfrag  = (_Float16*)alloc((size_t)PBH * 64 * FR * 2);
  _Float16* kfrag  = (_Float16*)alloc((size_t)PBH * 64 * FR * 2);
  _Float16* vfrag  = (_Float16*)alloc((size_t)PBH * 64 * FR * 2);
  float*    ctx    = (float*)   alloc((size_t)ROWS2 * AH_ * 4);
  _Float16* xfbuf  = (_Float16*)alloc((size_t)TOK * PAD * 2);
  _Float16* ybbuf  = (_Float16*)alloc((size_t)TOK * PAD * 2);
  _Float16* xffrag = (_Float16*)alloc((size_t)(TOK / 16) * KT_P * FR * 2);
  _Float16* ufrag  = (_Float16*)alloc((size_t)NL_ * KT_P * NT_P * FR * 2);
  _Float16* Tbuf   = (_Float16*)alloc((size_t)TOK * NL_ * PAD * 2);
  _Float16* tfrag  = (_Float16*)alloc((size_t)B_ * NL_ * 32 * KT_P * FR * 2);
  _Float16* ybfrag = (_Float16*)alloc((size_t)B_ * KT_P * 32 * FR * 2);

  auto blocks = [](int total, int bs) { return (total + bs - 1) / bs; };

  // 1) embedding -> X fragments (both shifted passes)
  k_pack_x<<<blocks(MT2 * KT_H * 32, 256), 256, 0, stream>>>(ids, emb, xfrag);
  // 2) weights -> B fragments
  k_pack_w<<<blocks(KT_H * NT_AH * 32, 256), 256, 0, stream>>>(Wq, wfq);
  k_pack_w<<<blocks(KT_H * NT_AH * 32, 256), 256, 0, stream>>>(Wk, wfk);
  k_pack_w<<<blocks(KT_H * NT_AH * 32, 256), 256, 0, stream>>>(Wv, wfv);
  // 8) U fragments (independent, overlapping the front of the pipe)
  k_pack_u<<<blocks(NL_ * KT_P * NT_P * 32, 256), 256, 0, stream>>>(U, ufrag);
  // 3) QKV GEMM (32x64 per wave)
  k_qkv<<<(MT2 / 2) * 3 * 12, 32, 0, stream>>>(xfrag, wfq, wfk, wfv, bq, bk, bv, qkv);
  // 4) RoPE + attention fragment packing
  k_pack_q<<<blocks(PBH * 32 * 2 * 32, 256), 256, 0, stream>>>(qkv, qfrag);
  k_pack_k<<<blocks(PBH * 32 * 2 * 32, 256), 256, 0, stream>>>(qkv, kfrag);
  k_pack_v<<<blocks(PBH * 16 * 4 * 32, 256), 256, 0, stream>>>(qkv, vfrag);
  // 5) attention
  k_attn<<<PBH * 32, 32, 0, stream>>>(qfrag, kfrag, vfrag, ctx);
  // 6) layernorm + ones/pad
  k_ln<<<ROWS2, 256, 0, stream>>>(ctx, lnw, lnb, xfbuf, ybbuf);
  // 7) xf -> A fragments
  k_pack_xf<<<blocks((TOK / 16) * KT_P * 32, 256), 256, 0, stream>>>(xfbuf, xffrag);
  // 9) T = xf @ U_o (32x32 per wave)
  k_bil1<<<(TOK / 32) * 12 * 25, 32, 0, stream>>>(xffrag, ufrag, Tbuf);
  // 10/11) repack for stage 2
  k_pack_t<<<blocks(B_ * NL_ * 32 * KT_P * 32, 256), 256, 0, stream>>>(Tbuf, tfrag);
  k_pack_yb<<<blocks(B_ * KT_P * 32 * 32, 256), 256, 0, stream>>>(ybbuf, ybfrag);
  // 12) out = T @ yb^T (64x64 per 4-wave block, async LDS staging)
  k_bil2<<<B_ * 12 * 8 * 8, 128, 0, stream>>>(tfrag, ybfrag, out);
}
